// MultiHeadedAttention_171798692217
// MI455X (gfx1250) — matrix operand order repaired and checked by reading, not compiled, not run on hardware
//
#include <hip/hip_runtime.h>
#include <hip/hip_bf16.h>

// ---------------------------------------------------------------------------
// Music-Transformer relative attention for MI455X (gfx1250).
// bf16 WMMA (v_wmma_f32_16x16x32_bf16) for all matmuls; flash-style fused
// attention so the S x S scores never touch HBM (256MB > 192MB L2; saves
// ~40us of HBM traffic at 23.3 TB/s). Tile staging uses the Tensor Data
// Mover (tensor_load_to_lds + s_wait_tensorcnt) where the tile is a plain
// bf16 2D block, and packed f32->bf16 conversion elsewhere.
// ---------------------------------------------------------------------------

typedef __attribute__((ext_vector_type(16))) __bf16 v16bf;
typedef __attribute__((ext_vector_type(8)))  __bf16 v8bf;
typedef __attribute__((ext_vector_type(4)))  __bf16 v4bf;
typedef __attribute__((ext_vector_type(8)))  float  v8f;
typedef __attribute__((ext_vector_type(4)))  float  v4fe;
typedef __attribute__((ext_vector_type(4)))  unsigned u32x4;
typedef __attribute__((ext_vector_type(8)))  int      i32x8;
typedef __attribute__((ext_vector_type(4)))  int      i32x4;

#define WMMA_BF16(A, B, C) \
  __builtin_amdgcn_wmma_f32_16x16x32_bf16(false, (A), false, (B), (short)0, (C), false, false)

constexpr int B_  = 4;
constexpr int S_  = 1024;
constexpr int D_  = 1024;
constexpr int H_  = 16;
constexpr int DH_ = 64;
constexpr int LDW = 72;   // LDS row stride (bf16): 64 data + 8 pad, 144B rows

// native fptrunc (RNE) -> v_cvt_pk_bf16_f32 / packed forms picked by backend
__device__ __forceinline__ __bf16 to_bf16(float f) { return (__bf16)f; }

// packed f32x4 -> bf16x4, 8B LDS store
__device__ __forceinline__ void cvt_store4(__bf16* dst, v4fe v) {
  *(v4bf*)dst = __builtin_convertvector(v, v4bf);
}

// ---------------------------------------------------------------------------
// Tensor Data Mover: DMA a rows x cols bf16 tile (row stride src_stride elems
// in global) into LDS with +8-elem row padding (pad 4 DWORDs every 32 DWORDs,
// matching LDW=72). D# per ISA 8.3/8.4: count=1, type=2, data_size=2B.
// Descriptor operands are block-uniform -> land in SGPRs. Issue from wave 0,
// then s_wait_tensorcnt 0 before the workgroup barrier.
// This toolchain carries the 6-arg builtin (amdgpu-toolchain / clang-23).
// ---------------------------------------------------------------------------
__device__ __forceinline__ void tdm_load_tile_bf16(unsigned lds_byte_addr,
                                                   const __bf16* gsrc,
                                                   unsigned rows, unsigned cols,
                                                   unsigned src_stride_elems) {
  unsigned long long ga = (unsigned long long)(const void*)gsrc;
  u32x4 g0;
  g0[0] = 1u;                                              // count=1, user D#
  g0[1] = lds_byte_addr;                                   // lds_addr
  g0[2] = (unsigned)ga;                                    // global_addr[31:0]
  g0[3] = (unsigned)((ga >> 32) & 0x01FFFFFFu) | (2u << 30); // [56:32] | type=2
  i32x8 g1;
  g1[0] = (int)((1u << 16)      // data_size = 2 bytes
              | (1u << 20)      // pad_enable
              | (4u << 22)      // pad_interval: 32 DWORDs (=128B row)
              | (3u << 25));    // pad_amount : 4 DWORDs  (=16B -> stride 72)
  g1[1] = (int)(cols << 16);                        // tensor_dim0[15:0]
  g1[2] = (int)((cols >> 16) | (rows << 16));       // dim0 hi | tensor_dim1 lo
  g1[3] = (int)((rows >> 16) | (cols << 16));       // dim1 hi | tile_dim0
  g1[4] = (int)rows;                                // tile_dim1 (tile_dim2=0)
  g1[5] = (int)src_stride_elems;                    // tensor_dim0_stride lo
  g1[6] = 0;
  g1[7] = 0;
  i32x4 z4 = {};
  i32x8 z8 = {};
  __builtin_amdgcn_tensor_load_to_lds(g0, g1, z4, z4, z8, 0);
}

// A-matrix fragment, 16x32 bf16 (ISA 7.12.2):
//  lanes 0-15 : row m=lane,    K = kb+{0..7}, kb+16+{0..7}
//  lanes 16-31: row m=lane-16, K = kb+8+{0..7}, kb+24+{0..7}
__device__ __forceinline__ v16bf load_frag_a(const __bf16* base, int ld, int lane, int kb) {
  int m = lane & 15, g = (lane >> 4) & 1;
  const __bf16* p = base + m * ld + kb + g * 8;
  v8bf lo = *(const v8bf*)p;
  v8bf hi = *(const v8bf*)(p + 16);
  return __builtin_shufflevector(lo, hi, 0,1,2,3,4,5,6,7,8,9,10,11,12,13,14,15);
}

// B-matrix fragment, 32x16 bf16; LDS holds B as [n][k] rows.
__device__ __forceinline__ v16bf load_frag_b(const __bf16* base, int ld, int lane, int kb) {
  int n = lane & 15, g = (lane >> 4) & 1;
  const __bf16* p = base + n * ld + kb + g * 8;
  v8bf lo = *(const v8bf*)p;
  v8bf hi = *(const v8bf*)(p + 16);
  return __builtin_shufflevector(lo, hi, 0,1,2,3,4,5,6,7,8,9,10,11,12,13,14,15);
}

// ---------------------------------------------------------------------------
// Projection GEMM: C = A(f32,[4096x1024]) @ W(f32,[1024x1024])^T -> bf16
// scattered head-major [b,h,s,dh]. fp32 source, so staging converts (packed).
// ---------------------------------------------------------------------------
__global__ __launch_bounds__(256)
void proj_gemm_kernel(const float* __restrict__ A, const float* __restrict__ W,
                      __bf16* __restrict__ out) {
  __shared__ __bf16 sA[64][LDW];
  __shared__ __bf16 sB[64][LDW];
  const int t = threadIdx.x, lane = t & 31, w = t >> 5;
  const int m0 = blockIdx.x * 64, n0 = blockIdx.y * 64;
  const int wm = (w >> 1) * 16, wn = (w & 1) * 32;
  v8f acc0 = {}, acc1 = {};

  for (int kk = 0; kk < D_; kk += 64) {
    __syncthreads();
    {
      int row = t >> 2, c0 = (t & 3) * 16;
      const v4fe* pa = (const v4fe*)(A + (size_t)(m0 + row) * D_ + kk + c0);
      const v4fe* pw = (const v4fe*)(W + (size_t)(n0 + row) * D_ + kk + c0);
      if (kk + 64 < D_) {               // global_prefetch_b8 for next K tile
        __builtin_prefetch(A + (size_t)(m0 + row) * D_ + kk + 64 + c0, 0, 1);
        __builtin_prefetch(W + (size_t)(n0 + row) * D_ + kk + 64 + c0, 0, 1);
      }
#pragma unroll
      for (int i = 0; i < 4; ++i) {
        cvt_store4(&sA[row][c0 + 4 * i], pa[i]);
        cvt_store4(&sB[row][c0 + 4 * i], pw[i]);
      }
    }
    __syncthreads();
#pragma unroll
    for (int ks = 0; ks < 64; ks += 32) {
      v16bf a  = load_frag_a(&sA[wm][0],      LDW, lane, ks);
      v16bf b0 = load_frag_b(&sB[wn][0],      LDW, lane, ks);
      v16bf b1 = load_frag_b(&sB[wn + 16][0], LDW, lane, ks);
      acc0 = WMMA_BF16(a, b0, acc0);
      acc1 = WMMA_BF16(a, b1, acc1);
    }
  }

  const int half = lane >> 4, nl = lane & 15;
#pragma unroll
  for (int r = 0; r < 8; ++r) {
    int m = m0 + wm + 8 * half + r;              // m = b*S + s
    int b = m >> 10, s = m & (S_ - 1);
#pragma unroll
    for (int tile = 0; tile < 2; ++tile) {
      int n = n0 + wn + tile * 16 + nl;          // n = h*DH + dh
      int h = n >> 6, dh = n & 63;
      float v = tile ? acc1[r] : acc0[r];
      out[(((size_t)(b * H_ + h)) * S_ + s) * DH_ + dh] = to_bf16(v);
    }
  }
}

// ---------------------------------------------------------------------------
// Output GEMM: d_out = Xbf(bf16) @ Wo(f32)^T, f32 out. A tile staged via TDM.
// ---------------------------------------------------------------------------
__global__ __launch_bounds__(256)
void out_gemm_kernel(const __bf16* __restrict__ A, const float* __restrict__ W,
                     float* __restrict__ out) {
  __shared__ __bf16 sA[64][LDW];
  __shared__ __bf16 sB[64][LDW];
  const int t = threadIdx.x, lane = t & 31, w = t >> 5;
  const int m0 = blockIdx.x * 64, n0 = blockIdx.y * 64;
  const int wm = (w >> 1) * 16, wn = (w & 1) * 32;
  v8f acc0 = {}, acc1 = {};
  const unsigned ldsA = (unsigned)(size_t)&sA[0][0];

  for (int kk = 0; kk < D_; kk += 64) {
    __syncthreads();                            // retire prior tile readers
    if (w == 0) {                               // async DMA: A tile -> LDS
      tdm_load_tile_bf16(ldsA, A + (size_t)m0 * D_ + kk, 64u, 64u, (unsigned)D_);
    }
    {                                            // stage W (f32 -> bf16 packed)
      int row = t >> 2, c0 = (t & 3) * 16;
      const v4fe* pw = (const v4fe*)(W + (size_t)(n0 + row) * D_ + kk + c0);
#pragma unroll
      for (int i = 0; i < 4; ++i) cvt_store4(&sB[row][c0 + 4 * i], pw[i]);
    }
    if (w == 0) __builtin_amdgcn_s_wait_tensorcnt(0);
    __syncthreads();
#pragma unroll
    for (int ks = 0; ks < 64; ks += 32) {
      v16bf a  = load_frag_a(&sA[wm][0],      LDW, lane, ks);
      v16bf b0 = load_frag_b(&sB[wn][0],      LDW, lane, ks);
      v16bf b1 = load_frag_b(&sB[wn + 16][0], LDW, lane, ks);
      acc0 = WMMA_BF16(a, b0, acc0);
      acc1 = WMMA_BF16(a, b1, acc1);
    }
  }

  const int half = lane >> 4, nl = lane & 15;
#pragma unroll
  for (int r = 0; r < 8; ++r) {
    int m = m0 + wm + 8 * half + r;
#pragma unroll
    for (int tile = 0; tile < 2; ++tile) {
      int n = n0 + wn + tile * 16 + nl;
      out[(size_t)m * D_ + n] = tile ? acc1[r] : acc0[r];
    }
  }
}

// ---------------------------------------------------------------------------
// Fused relative-position flash attention. Grid: (S/64, B*H), 128 thr/4 waves.
// Wave owns 16 query rows -> softmax row stats reduce with shfl_xor inside a
// 16-lane half. Skew: stage PE rows r = S-1+j0-i0-63..+64 (TDM-unfriendly due
// to OOB zeroing), compute pos = Q @ PE^T over two extra 16-wide windows per
// score subtile, gather the diagonal with ds_bpermute, add where j<=i.
// Q and K tiles are contiguous bf16 blocks -> staged with the TDM.
// ---------------------------------------------------------------------------
__global__ __launch_bounds__(128)
void attn_kernel(const __bf16* __restrict__ Q, const __bf16* __restrict__ K,
                 const __bf16* __restrict__ V, const float* __restrict__ pe,
                 const int* __restrict__ msk, __bf16* __restrict__ X) {
  __shared__ __bf16 sQ[64][LDW];
  __shared__ __bf16 sK[64][LDW];
  __shared__ __bf16 sVt[64][LDW];     // V transposed: sVt[d][j]
  __shared__ __bf16 sPE[128][LDW];    // relative-position band
  __shared__ __bf16 sP[64][LDW];      // softmax probabilities (bf16)
  __shared__ float  sMk[64];
  __shared__ float  sMq[64];

  const int t = threadIdx.x, lane = t & 31, w = t >> 5;
  const int i0 = blockIdx.x * 64;
  const int bh = blockIdx.y, b = bh >> 4, h = bh & 15;
  const int half = lane >> 4, nl = lane & 15;
  const int wr0 = w * 16;

  const __bf16* Qh = Q + (size_t)bh * S_ * DH_;
  const __bf16* Kh = K + (size_t)bh * S_ * DH_;
  const __bf16* Vh = V + (size_t)bh * S_ * DH_;

  if (w == 0) {   // Q tile (64x64 bf16, contiguous rows) via Tensor Data Mover
    tdm_load_tile_bf16((unsigned)(size_t)&sQ[0][0],
                       Qh + (size_t)i0 * DH_, 64u, 64u, (unsigned)DH_);
  }
  if (t < 64) sMq[t] = (float)msk[(size_t)b * S_ + i0 + t];

  float m_r[8], l_r[8];
  v8f acc_o[4] = {};
#pragma unroll
  for (int r = 0; r < 8; ++r) { m_r[r] = -3.0e38f; l_r[r] = 0.0f; }

  for (int j0 = 0; j0 < S_; j0 += 64) {
    __syncthreads();                              // retire prior tile readers
    if (w == 0) {                                 // K tile via TDM
      tdm_load_tile_bf16((unsigned)(size_t)&sK[0][0],
                         Kh + (size_t)j0 * DH_, 64u, 64u, (unsigned)DH_);
    }
    {   // stage V^T and key mask (VALU path: transpose)
      int row = t >> 1, c0 = (t & 1) * 32;
      const __bf16* pv = Vh + (size_t)(j0 + row) * DH_ + c0;
#pragma unroll
      for (int i = 0; i < 32; ++i) sVt[c0 + i][row] = pv[i];
      if (t < 64) sMk[t] = (float)msk[(size_t)b * S_ + j0 + t];
    }
    {   // stage PE band: rr in [0,128), r = S-1+j0-i0-63+rr, zero when OOB
      int rr = t;
      int r = (S_ - 1 + j0 - i0 - 63) + rr;
      if (r >= 0 && r < S_) {
        const v4fe* p = (const v4fe*)(pe + ((size_t)h * S_ + r) * DH_);
#pragma unroll
        for (int i = 0; i < 16; ++i) cvt_store4(&sPE[rr][4 * i], p[i]);
      } else {
        v4bf z = {};
#pragma unroll
        for (int c = 0; c < 16; ++c) *(v4bf*)&sPE[rr][4 * c] = z;
      }
    }
    if (w == 0) __builtin_amdgcn_s_wait_tensorcnt(0);   // Q (1st iter) + K done
    __syncthreads();

    float sv[4][8];
#pragma unroll
    for (int jn4 = 0; jn4 < 4; ++jn4) {
      const int jn = jn4 * 16;
      const int rro = jn - wr0 + 48;                   // in [0,96]
      v8f qkacc = {}, p0 = {}, p1 = {};
#pragma unroll
      for (int ks = 0; ks < 64; ks += 32) {
        v16bf a   = load_frag_a(&sQ[wr0][0],       LDW, lane, ks);
        v16bf bk  = load_frag_b(&sK[jn][0],        LDW, lane, ks);
        v16bf bp0 = load_frag_b(&sPE[rro][0],      LDW, lane, ks);
        v16bf bp1 = load_frag_b(&sPE[rro + 16][0], LDW, lane, ks);
        qkacc = WMMA_BF16(a, bk,  qkacc);
        p0    = WMMA_BF16(a, bp0, p0);
        p1    = WMMA_BF16(a, bp1, p1);
      }
#pragma unroll
      for (int r = 0; r < 8; ++r) {
        int brow = 8 * half + r;
        int off = nl - brow + 15;                      // 0..30: skew gather
        int srclane = (half << 4) | (off & 15);
        int g0 = __builtin_amdgcn_ds_bpermute(srclane << 2, __builtin_bit_cast(int, p0[r]));
        int g1 = __builtin_amdgcn_ds_bpermute(srclane << 2, __builtin_bit_cast(int, p1[r]));
        float posv = __builtin_bit_cast(float, (off < 16) ? g0 : g1);
        float v = qkacc[r] * 0.125f;                   // DH^-0.5
        int jg = j0 + jn + nl, ig = i0 + wr0 + brow;
        if (jg <= ig) v += posv * 8.0f;                // DH^0.5, zero for j>i
        if (sMk[jn + nl] * sMq[wr0 + brow] == 0.0f) v = -1.0e9f;
        sv[jn4][r] = v;
      }
    }

    // online softmax (wave-local 16-lane shfl reductions)
#pragma unroll
    for (int r = 0; r < 8; ++r) {
      int brow = 8 * half + r;
      float mx = fmaxf(fmaxf(sv[0][r], sv[1][r]), fmaxf(sv[2][r], sv[3][r]));
#pragma unroll
      for (int o = 8; o >= 1; o >>= 1) mx = fmaxf(mx, __shfl_xor(mx, o, 32));
      float mn = fmaxf(m_r[r], mx);
      float corr = __expf(m_r[r] - mn);
      m_r[r] = mn;
      float ps = 0.0f;
#pragma unroll
      for (int tt = 0; tt < 4; ++tt) {
        float p = __expf(sv[tt][r] - mn);
        ps += p;
        sP[wr0 + brow][tt * 16 + nl] = to_bf16(p);
      }
#pragma unroll
      for (int o = 8; o >= 1; o >>= 1) ps += __shfl_xor(ps, o, 32);
      l_r[r] = l_r[r] * corr + ps;
#pragma unroll
      for (int tt = 0; tt < 4; ++tt) acc_o[tt][r] = acc_o[tt][r] * corr;
    }

    // acc_o += P @ V  (P rows wave-private; same-wave LDS dep via dscnt)
#pragma unroll
    for (int ks = 0; ks < 64; ks += 32) {
      v16bf a = load_frag_a(&sP[wr0][0], LDW, lane, ks);
#pragma unroll
      for (int nd = 0; nd < 4; ++nd) {
        v16bf bv = load_frag_b(&sVt[nd * 16][0], LDW, lane, ks);
        acc_o[nd] = WMMA_BF16(a, bv, acc_o[nd]);
      }
    }
  }

  // X[b, s, h*DH + d] = acc / l   (bf16, feeds output GEMM)
#pragma unroll
  for (int r = 0; r < 8; ++r) {
    int srow = i0 + wr0 + 8 * half + r;
    float inv = 1.0f / l_r[r];
#pragma unroll
    for (int nd = 0; nd < 4; ++nd) {
      int d = nd * 16 + nl;
      X[((size_t)b * S_ + srow) * D_ + h * DH_ + d] = to_bf16(acc_o[nd][r] * inv);
    }
  }
}

// ---------------------------------------------------------------------------
extern "C" void kernel_launch(void* const* d_in, const int* in_sizes, int n_in,
                              void* d_out, int out_size, void* d_ws, size_t ws_size,
                              hipStream_t stream) {
  (void)in_sizes; (void)n_in; (void)out_size; (void)ws_size;
  const float* query = (const float*)d_in[0];
  const float* key   = (const float*)d_in[1];
  const float* value = (const float*)d_in[2];
  const int*   mask  = (const int*)  d_in[3];
  const float* pos   = (const float*)d_in[4];
  const float* Wq    = (const float*)d_in[5];
  const float* Wk    = (const float*)d_in[6];
  const float* Wv    = (const float*)d_in[7];
  const float* Wo    = (const float*)d_in[8];
  float* out = (float*)d_out;

  char* ws = (char*)d_ws;
  const size_t seg = (size_t)B_ * H_ * S_ * DH_ * sizeof(__bf16);   // 8 MB
  __bf16* Qbf = (__bf16*)(ws);
  __bf16* Kbf = (__bf16*)(ws + seg);
  __bf16* Vbf = (__bf16*)(ws + 2 * seg);
  __bf16* Xbf = (__bf16*)(ws + 3 * seg);

  dim3 gg(B_ * S_ / 64, D_ / 64);   // (64, 16)
  proj_gemm_kernel<<<gg, 256, 0, stream>>>(query, Wq, Qbf);
  proj_gemm_kernel<<<gg, 256, 0, stream>>>(key,   Wk, Kbf);
  proj_gemm_kernel<<<gg, 256, 0, stream>>>(value, Wv, Vbf);
  attn_kernel<<<dim3(S_ / 64, B_ * H_), 128, 0, stream>>>(Qbf, Kbf, Vbf, pos, mask, Xbf);
  out_gemm_kernel<<<gg, 256, 0, stream>>>(Xbf, Wo, out);
}
